// TMIModule_62672162783422
// MI455X (gfx1250) — 8-bit matrix operand order repaired and checked by reading, compiled, not run on hardware
//
#include <hip/hip_runtime.h>
#include <hip/hip_bf16.h>

#define BB   16
#define TT   8
#define CC   32
#define HH   64
#define WW   64
#define HW   (HH*WW)
#define CHW  (CC*HW)
#define TCHW (TT*CHW)
#define BC   (BB*CC)
#define TILE 32
#define SSTR 516

typedef __attribute__((ext_vector_type(8))) int v8i;

__device__ __forceinline__ unsigned onehot01(unsigned u) {
    return (~(((u & 0x7F7F7F7Fu) + 0x7F7F7F7Fu) | u | 0x7F7F7F7Fu)) >> 7;
}

__device__ __forceinline__ float xlogx(float c) {
    return (c > 0.f) ? c * __logf(c) : 0.f;
}

__global__ void tmi_zero_kernel(unsigned* p, int n) {
    int i = blockIdx.x * blockDim.x + threadIdx.x;
    if (i < n) p[i] = 0u;
}

__global__ __launch_bounds__(256)
void tmi_hist_kernel(const float* __restrict__ x,
                     unsigned* __restrict__ glob_joint,
                     float* __restrict__ ele) {
    __shared__ __align__(16) unsigned char s0buf[TILE * SSTR];
    __shared__ __align__(16) unsigned char stbuf[TILE * SSTR];
    __shared__ unsigned globacc[256];

    const int tid = threadIdx.x;
    const int hw0 = blockIdx.x * TILE;
    const int t   = blockIdx.y;

    globacc[tid] = 0u;

    for (int i4 = tid; i4 < (BC * TILE) / 4; i4 += 256) {
        int e    = i4 << 2;
        int bc   = e >> 5;
        int off0 = e & (TILE - 1);
        int bb = bc >> 5;
        int cc = bc & (CC - 1);
        size_t g = (size_t)bb * TCHW + (size_t)cc * HW + (size_t)hw0 + off0;
        float4 va = *(const float4*)(x + g);
        float4 vb = *(const float4*)(x + g + (size_t)t * CHW);
        s0buf[(off0 + 0) * SSTR + bc] = (unsigned char)(int)va.x;
        s0buf[(off0 + 1) * SSTR + bc] = (unsigned char)(int)va.y;
        s0buf[(off0 + 2) * SSTR + bc] = (unsigned char)(int)va.z;
        s0buf[(off0 + 3) * SSTR + bc] = (unsigned char)(int)va.w;
        stbuf[(off0 + 0) * SSTR + bc] = (unsigned char)(int)vb.x;
        stbuf[(off0 + 1) * SSTR + bc] = (unsigned char)(int)vb.y;
        stbuf[(off0 + 2) * SSTR + bc] = (unsigned char)(int)vb.z;
        stbuf[(off0 + 3) * SSTR + bc] = (unsigned char)(int)vb.w;
    }
    __syncthreads();

    const int wv   = tid >> 5;
    const int lane = tid & 31;
    const int m    = lane & 15;
    const int half = lane >> 4;
    const unsigned mrep = (unsigned)m * 0x01010101u;

    for (int q = 0; q < 4; ++q) {
        const int off = wv * 4 + q;
        const unsigned char* aS = s0buf + off * SSTR;
        const unsigned char* bS = stbuf + off * SSTR;

        v8i acc0 = {0, 0, 0, 0, 0, 0, 0, 0};
        v8i acc1 = {0, 0, 0, 0, 0, 0, 0, 0};
        #pragma unroll
        for (int ch = 0; ch < 8; ch += 2) {
            {
                const int kb = ch * 64;
                v8i A, B;
                #pragma unroll
                for (int v = 0; v < 8; ++v) {
                    int ka  = kb + ((v >> 2) * 32 + (v & 2) * 8 + (v & 1) * 4 + half * 8);
                    int kbB = kb + ((v >> 2) * 32 + (v & 2) * 8 + (v & 1) * 4 + half * 8);
                    unsigned wa = *(const unsigned*)(aS + ka) ^ mrep;
                    unsigned wb = *(const unsigned*)(bS + kbB) ^ mrep;
                    A[v] = (int)onehot01(wa);
                    B[v] = (int)onehot01(wb);
                }
                acc0 = __builtin_amdgcn_wmma_i32_16x16x64_iu8(
                    false, A, false, B, acc0, false, false);
            }
            {
                const int kb = (ch + 1) * 64;
                v8i A, B;
                #pragma unroll
                for (int v = 0; v < 8; ++v) {
                    int ka  = kb + ((v >> 2) * 32 + (v & 2) * 8 + (v & 1) * 4 + half * 8);
                    int kbB = kb + ((v >> 2) * 32 + (v & 2) * 8 + (v & 1) * 4 + half * 8);
                    unsigned wa = *(const unsigned*)(aS + ka) ^ mrep;
                    unsigned wb = *(const unsigned*)(bS + kbB) ^ mrep;
                    A[v] = (int)onehot01(wa);
                    B[v] = (int)onehot01(wb);
                }
                acc1 = __builtin_amdgcn_wmma_i32_16x16x64_iu8(
                    false, A, false, B, acc1, false, false);
            }
        }

        float sj = 0.f, sr = 0.f;
        int pc = 0;
        int jr[8];
        #pragma unroll
        for (int r = 0; r < 8; ++r) {
            int cv = acc0[r] + acc1[r];
            jr[r] = cv;
            sj += xlogx((float)cv);
            int rs = cv;
            rs += __shfl_xor(rs, 1);
            rs += __shfl_xor(rs, 2);
            rs += __shfl_xor(rs, 4);
            rs += __shfl_xor(rs, 8);
            sr += xlogx((float)rs);
            pc += cv;
        }
        int cs = pc + __shfl_xor(pc, 16);
        float qv = sj - sr * 0.0625f - xlogx((float)cs) * 0.5f;
        #pragma unroll
        for (int d = 1; d < 32; d <<= 1) qv += __shfl_xor(qv, d);
        if (lane == 0)
            ele[(size_t)t * HW + hw0 + off] =
                (qv + 512.f * __logf(512.f)) * (1.f / 512.f);

        #pragma unroll
        for (int r = 0; r < 8; ++r)
            atomicAdd(&globacc[(r + 8 * half) * 16 + m], (unsigned)jr[r]);
    }
    __syncthreads();
    atomicAdd(&glob_joint[t * 256 + tid], globacc[tid]);
}

__global__ void tmi_prob_kernel(const unsigned* __restrict__ glob_joint,
                                const float* __restrict__ ele,
                                const int* __restrict__ epoch,
                                float* __restrict__ prob) {
    __shared__ float gl[TT];
    int tid = threadIdx.x;
    if (tid < TT) {
        double rows[16], cols[16];
        for (int i = 0; i < 16; ++i) { rows[i] = 0.0; cols[i] = 0.0; }
        double sj = 0.0, n = 0.0;
        for (int i = 0; i < 256; ++i) {
            double cv = (double)glob_joint[tid * 256 + i];
            n += cv;
            if (cv > 0.0) sj += cv * log(cv);
            rows[i >> 4] += cv;
            cols[i & 15] += cv;
        }
        double s = sj;
        for (int i = 0; i < 16; ++i) {
            if (rows[i] > 0.0) s -= rows[i] * log(rows[i]);
            if (cols[i] > 0.0) s -= cols[i] * log(cols[i]);
        }
        s += n * log(n);
        gl[tid] = (float)(s / n);
    }
    __syncthreads();
    float esc = (float)(*epoch) * (1.f / 200.f);
    for (int i = tid; i < TT * HW; i += blockDim.x) {
        int t  = i >> 12;
        int hw = i & (HW - 1);
        float p = 0.f;
        if (t != 0) {
            float v = (ele[i] * gl[t]) / (ele[hw] * gl[0]) * esc;
            v = fmaxf(v, 0.f);
            p = fminf(v, 1.f);
        }
        prob[i] = p;
    }
}

__global__ __launch_bounds__(256)
void tmi_mask_kernel(const float* __restrict__ x,
                     const float* __restrict__ noise,
                     const float* __restrict__ prob,
                     float* __restrict__ out) {
    size_t base = ((size_t)blockIdx.x * blockDim.x + threadIdx.x) * 4;
    int t  = (int)((base >> 17) & (TT - 1));
    int hw = (int)(base & (HW - 1));
    float4 xv = *(const float4*)(x + base);
    float4 nv = *(const float4*)(noise + base);
    const float* pr = prob + t * HW + hw;
    float4 ov;
    ov.x = (nv.x < pr[0]) ? 0.f : xv.x;
    ov.y = (nv.y < pr[1]) ? 0.f : xv.y;
    ov.z = (nv.z < pr[2]) ? 0.f : xv.z;
    ov.w = (nv.w < pr[3]) ? 0.f : xv.w;
    *(float4*)(out + base) = ov;
}

extern "C" void kernel_launch(void* const* d_in, const int* in_sizes, int n_in,
                              void* d_out, int out_size, void* d_ws, size_t ws_size,
                              hipStream_t stream) {
    (void)in_sizes; (void)n_in; (void)out_size; (void)ws_size;
    const float* x     = (const float*)d_in[0];
    const float* noise = (const float*)d_in[1];
    const int*   epoch = (const int*)d_in[2];
    float* out = (float*)d_out;

    unsigned* glob_joint = (unsigned*)d_ws;
    float* ele  = (float*)((char*)d_ws + 8192);
    float* prob = (float*)((char*)d_ws + 8192 + TT * HW * sizeof(float));

    tmi_zero_kernel<<<8, 256, 0, stream>>>(glob_joint, TT * 256);
    tmi_hist_kernel<<<dim3(HW / TILE, TT), 256, 0, stream>>>(x, glob_joint, ele);
    tmi_prob_kernel<<<1, 256, 0, stream>>>(glob_joint, ele, epoch, prob);
    tmi_mask_kernel<<<(BB * TT * CC * HW) / 4 / 256, 256, 0, stream>>>(x, noise, prob, out);
}
